// PointGroup_31748398252316
// MI455X (gfx1250) — matrix operand order repaired and checked by reading, not compiled, not run on hardware
//
#include <hip/hip_runtime.h>
#include <hip/hip_bf16.h>

// ---------------- problem constants ----------------
#define N_PTS  262144
#define C_DIM  256
#define K_CLS  20
#define TM     64          // rows per block
#define KSTEP  32          // WMMA K per step (bf16)
#define EPSF   1e-8f
#define BNEPS  1e-3f

// workspace float layout (in d_ws, floats)
enum {
  WS_SUM    = 0,           // [256] per-channel sum of h
  WS_SUM2   = 256,         // [256] per-channel sum of h^2
  WS_SCALE  = 512,         // [256] gamma*rstd
  WS_SHIFT  = 768,         // [256] beta - mu*gamma*rstd
  WS_SEGSUM = 1024,
  WS_SEGCNT = 1025,
  WS_L1     = 1026,
  WS_COS    = 1027,
  WS_MASK   = 1028,
  WS_TOTAL  = 1029
};

// bf16 weight images in d_ws (byte offsets, all 16B aligned)
#define W1T_OFF  8192                       // 8 slices x [256][32] bf16 = 128 KB
#define W1T_SLICE_BYTES 16384
#define WST_OFF  (W1T_OFF + 8 * W1T_SLICE_BYTES)   // [32][256] bf16 = 16 KB
#define W2T_OFF  (WST_OFF + 16384)                 // [16][256] bf16 = 8 KB

typedef __attribute__((ext_vector_type(16))) __bf16        v16bf;
typedef __attribute__((ext_vector_type(8)))  float         v8f;
typedef __attribute__((ext_vector_type(4)))  unsigned int  u32x4;

union Frag16 { u32x4 u[2]; v16bf v; };

// ---- CDNA5 async copy: global -> LDS, 16B per lane, ASYNCcnt tracked ----
__device__ inline void async_copy_b128(const void* gbase, unsigned byteOff,
                                       unsigned ldsAddr) {
  // GVS mode: mem_addr = SADDR + VADDR; VDST supplies per-lane LDS address.
  asm volatile("global_load_async_to_lds_b128 %0, %1, %2"
               :: "v"(ldsAddr), "v"(byteOff), "s"(gbase) : "memory");
}
__device__ inline void wait_async0() {
  asm volatile("s_wait_asynccnt 0" ::: "memory");
}
__device__ inline unsigned lds_addr32(const void* p) {
  return (unsigned)(size_t)p;   // flat LDS address: offset lives in addr[31:0]
}

// A fragment (16-bit A, 16x32): lane m = lane&15, khalf = lane>>4, K = 16*khalf + s.
__device__ inline v16bf load_afrag(const __bf16* rowPtr, int kk, int khalf) {
  Frag16 f;
  f.u[0] = *(const u32x4*)(rowPtr + kk + khalf * 16);
  f.u[1] = *(const u32x4*)(rowPtr + kk + khalf * 16 + 8);
  return f.v;
}
// B fragment (32x16) from [n][k0..31] staging: lanes 0-15 K=0..15 of col n,
// lanes 16-31 K=16..31.
__device__ inline v16bf load_bfrag(const __bf16* colPtr, int khalf) {
  Frag16 f;
  f.u[0] = *(const u32x4*)(colPtr + khalf * 16);
  f.u[1] = *(const u32x4*)(colPtr + khalf * 16 + 8);
  return f.v;
}

__device__ inline v8f wmma_bf16(v16bf a, v16bf b, v8f c) {
  return __builtin_amdgcn_wmma_f32_16x16x32_bf16(
      false, a, false, b, (short)0, c, false, false);
}

__device__ inline float wave_reduce_add(float v) {
  #pragma unroll
  for (int off = 16; off > 0; off >>= 1) v += __shfl_xor(v, off, 32);
  return v;
}

// ---------------- init: zero accumulators ----------------
__global__ void pg_init(float* __restrict__ g) {
  for (int i = threadIdx.x; i < WS_TOTAL; i += blockDim.x) g[i] = 0.f;
}

// ---------------- prep: pre-convert & swizzle weights to bf16 images ----------------
__global__ __launch_bounds__(256) void pg_prep(
    const float* __restrict__ w1, const float* __restrict__ wsW,
    const float* __restrict__ w2, char* __restrict__ wsbase) {
  __bf16* w1t = (__bf16*)(wsbase + W1T_OFF);   // [slice][n][k]
  __bf16* wst = (__bf16*)(wsbase + WST_OFF);   // [n][k]
  __bf16* w2t = (__bf16*)(wsbase + W2T_OFF);   // [n][k]
  int gid = blockIdx.x * 256 + threadIdx.x;
  int stride = gridDim.x * 256;
  for (int i = gid; i < 8 * 256 * 32; i += stride) {
    int s = i >> 13, n = (i >> 5) & 255, k = i & 31;
    w1t[i] = (__bf16)w1[(size_t)(s * 32 + k) * C_DIM + n];
  }
  for (int i = gid; i < 32 * 256; i += stride) {
    int n = i >> 8, k = i & 255;
    wst[i] = (__bf16)((n < K_CLS) ? wsW[k * K_CLS + n] : 0.0f);
  }
  for (int i = gid; i < 16 * 256; i += stride) {
    int n = i >> 8, k = i & 255;
    w2t[i] = (__bf16)((n < 3) ? w2[k * 3 + n] : 0.0f);
  }
}

// ---------------- pass1: GEMM1 stats + logits + CE ----------------
__global__ __launch_bounds__(256) void pg_pass1(
    const float* __restrict__ feat, const int* __restrict__ segment,
    const float* __restrict__ b1,   const float* __restrict__ bsB,
    char* __restrict__ wsbase)
{
  __shared__ __bf16 sFeat[TM][C_DIM];      // 32 KB  feat tile (bf16)
  __shared__ __bf16 sB[C_DIM][KSTEP];      // 16 KB  w1 K-slice, [n][k]
  __shared__ __bf16 sWs[32][C_DIM];        // 16 KB  seg weights, [n][k]
  __shared__ float  sLogits[TM][32];       //  8 KB
  __shared__ float  sSum[C_DIM], sSum2[C_DIM];

  float* g_acc = (float*)wsbase;
  const int tid   = threadIdx.x;
  const int lane  = tid & 31;
  const int wave  = tid >> 5;
  const int khalf = lane >> 4;
  const int rowBase = blockIdx.x * TM;

  // async-copy seg-head weights (pre-swizzled bf16) into LDS
  {
    unsigned base = lds_addr32(&sWs[0][0]);
    #pragma unroll
    for (int i = 0; i < 4; ++i) {
      unsigned off = (unsigned)(i * 256 + tid) * 16u;
      async_copy_b128(wsbase + WST_OFF, off, base + off);
    }
  }
  // stage feat tile as bf16 (coalesced fp32 reads; conversion needs VALU)
  for (int i = tid; i < TM * C_DIM; i += 256) {
    int m = i >> 8, c = i & 255;
    sFeat[m][c] = (__bf16)feat[(size_t)(rowBase + m) * C_DIM + c];
  }
  if (tid < C_DIM) { sSum[tid] = 0.f; sSum2[tid] = 0.f; }
  wait_async0();
  __syncthreads();

  // ---- GEMM1: h = feat @ w1 (need only column stats of h) ----
  const int mt     = wave >> 1;
  const int ntBase = (wave & 1) * 8;
  const int mrow   = mt * 16 + (lane & 15);
  const unsigned sBbase = lds_addr32(&sB[0][0]);

  v8f acc[8];
  #pragma unroll
  for (int j = 0; j < 8; ++j) acc[j] = {};

  for (int kk = 0; kk < C_DIM; kk += KSTEP) {
    __syncthreads();  // previous slice consumed
    const char* slice = wsbase + W1T_OFF + (kk >> 5) * W1T_SLICE_BYTES;
    #pragma unroll
    for (int i = 0; i < 4; ++i) {
      unsigned off = (unsigned)(i * 256 + tid) * 16u;
      async_copy_b128(slice, off, sBbase + off);
    }
    wait_async0();
    __syncthreads();
    v16bf a = load_afrag(&sFeat[mrow][0], kk, khalf);
    #pragma unroll
    for (int j = 0; j < 8; ++j) {
      int n = (ntBase + j) * 16 + (lane & 15);
      v16bf b = load_bfrag(&sB[n][0], khalf);
      acc[j] = wmma_bf16(a, b, acc[j]);
    }
  }

  // ---- add b1, accumulate column sum / sumsq ----
  #pragma unroll
  for (int j = 0; j < 8; ++j) {
    int col = (ntBase + j) * 16 + (lane & 15);
    float bias = b1[col];
    float s = 0.f, s2 = 0.f;
    #pragma unroll
    for (int r = 0; r < 8; ++r) {
      float v = acc[j][r] + bias;
      s += v; s2 += v * v;
    }
    s  += __shfl_xor(s, 16, 32);
    s2 += __shfl_xor(s2, 16, 32);
    if (lane < 16) { atomicAdd(&sSum[col], s); atomicAdd(&sSum2[col], s2); }
  }
  __syncthreads();
  if (tid < C_DIM) {
    atomicAdd(&g_acc[WS_SUM  + tid], sSum[tid]);
    atomicAdd(&g_acc[WS_SUM2 + tid], sSum2[tid]);
  }

  // ---- logits = feat @ ws (padded to 32 cols), one tile per wave ----
  const int mt2 = wave & 3;
  const int nt2 = wave >> 2;
  v8f accL = {};
  #pragma unroll
  for (int kk = 0; kk < C_DIM; kk += KSTEP) {
    v16bf a = load_afrag(&sFeat[mt2 * 16 + (lane & 15)][0], kk, khalf);
    int n = nt2 * 16 + (lane & 15);
    Frag16 f;
    f.u[0] = *(const u32x4*)(&sWs[n][kk + khalf * 16]);
    f.u[1] = *(const u32x4*)(&sWs[n][kk + khalf * 16 + 8]);
    accL = wmma_bf16(a, f.v, accL);
  }
  {
    int col = nt2 * 16 + (lane & 15);
    float bias = (col < K_CLS) ? bsB[col] : 0.f;
    #pragma unroll
    for (int r = 0; r < 8; ++r) {
      int m = mt2 * 16 + khalf * 8 + r;
      sLogits[m][col] = accL[r] + bias;
    }
  }
  __syncthreads();

  // ---- per-row CE with ignore mask (waves 0,1 fully active) ----
  if (tid < TM) {
    int row = rowBase + tid;
    int seg = segment[row];
    bool valid = (seg != -1);
    int tgt = valid ? seg : 0;
    float mx = -3.4e38f;
    #pragma unroll
    for (int j = 0; j < K_CLS; ++j) mx = fmaxf(mx, sLogits[tid][j]);
    float se = 0.f;
    #pragma unroll
    for (int j = 0; j < K_CLS; ++j) se += expf(sLogits[tid][j] - mx);
    float nll = (logf(se) + mx) - sLogits[tid][tgt];
    float c0 = wave_reduce_add(valid ? nll : 0.f);
    float c1 = wave_reduce_add(valid ? 1.f : 0.f);
    if (lane == 0) {
      atomicAdd(&g_acc[WS_SEGSUM], c0);
      atomicAdd(&g_acc[WS_SEGCNT], c1);
    }
  }
}

// ---------------- pass2: finalize BN scale/shift ----------------
__global__ void pg_pass2(const float* __restrict__ gamma,
                         const float* __restrict__ beta,
                         float* __restrict__ g_acc) {
  int c = threadIdx.x;  // 256 threads
  float inv_n = 1.0f / (float)N_PTS;
  float mu  = g_acc[WS_SUM + c] * inv_n;
  float var = g_acc[WS_SUM2 + c] * inv_n - mu * mu;
  float sc  = gamma[c] * rsqrtf(var + BNEPS);
  g_acc[WS_SCALE + c] = sc;
  g_acc[WS_SHIFT + c] = beta[c] - mu * sc;
}

// ---------------- pass3: recompute h, BN+ReLU, bias head, L1+cos ----------------
__global__ __launch_bounds__(256) void pg_pass3(
    const float* __restrict__ feat,  const float* __restrict__ coord,
    const float* __restrict__ centroid, const int* __restrict__ instance,
    const float* __restrict__ b1,    const float* __restrict__ b2,
    char* __restrict__ wsbase)
{
  __shared__ __bf16 sFeat[TM][C_DIM];      // feat tile, later normalized h
  __shared__ __bf16 sB[C_DIM][KSTEP];
  __shared__ __bf16 sW2[16][C_DIM];        // w2 padded 3->16, [n][k]
  __shared__ float  sBias[TM][4];

  float* g_acc = (float*)wsbase;
  const int tid   = threadIdx.x;
  const int lane  = tid & 31;
  const int wave  = tid >> 5;
  const int khalf = lane >> 4;
  const int rowBase = blockIdx.x * TM;

  {
    unsigned base = lds_addr32(&sW2[0][0]);
    #pragma unroll
    for (int i = 0; i < 2; ++i) {
      unsigned off = (unsigned)(i * 256 + tid) * 16u;
      async_copy_b128(wsbase + W2T_OFF, off, base + off);
    }
  }
  for (int i = tid; i < TM * C_DIM; i += 256) {
    int m = i >> 8, c = i & 255;
    sFeat[m][c] = (__bf16)feat[(size_t)(rowBase + m) * C_DIM + c];
  }
  wait_async0();
  __syncthreads();

  // ---- recompute GEMM1 ----
  const int mt     = wave >> 1;
  const int ntBase = (wave & 1) * 8;
  const int mrow   = mt * 16 + (lane & 15);
  const unsigned sBbase = lds_addr32(&sB[0][0]);

  v8f acc[8];
  #pragma unroll
  for (int j = 0; j < 8; ++j) acc[j] = {};

  for (int kk = 0; kk < C_DIM; kk += KSTEP) {
    __syncthreads();
    const char* slice = wsbase + W1T_OFF + (kk >> 5) * W1T_SLICE_BYTES;
    #pragma unroll
    for (int i = 0; i < 4; ++i) {
      unsigned off = (unsigned)(i * 256 + tid) * 16u;
      async_copy_b128(slice, off, sBbase + off);
    }
    wait_async0();
    __syncthreads();
    v16bf a = load_afrag(&sFeat[mrow][0], kk, khalf);
    #pragma unroll
    for (int j = 0; j < 8; ++j) {
      int n = (ntBase + j) * 16 + (lane & 15);
      v16bf b = load_bfrag(&sB[n][0], khalf);
      acc[j] = wmma_bf16(a, b, acc[j]);
    }
  }

  // ---- +b1, BN, ReLU, write normalized bf16 tile back over sFeat ----
  __syncthreads();
  #pragma unroll
  for (int j = 0; j < 8; ++j) {
    int col  = (ntBase + j) * 16 + (lane & 15);
    float bias  = b1[col];
    float scale = g_acc[WS_SCALE + col];
    float shift = g_acc[WS_SHIFT + col];
    #pragma unroll
    for (int r = 0; r < 8; ++r) {
      int m = mt * 16 + khalf * 8 + r;
      float v = fmaxf((acc[j][r] + bias) * scale + shift, 0.f);
      sFeat[m][col] = (__bf16)v;
    }
  }
  __syncthreads();

  // ---- GEMM2: bias_pred = relu(hn) @ w2 (padded), waves 0..3 ----
  if (wave < 4) {
    const int mtb = wave;
    v8f a2 = {};
    #pragma unroll
    for (int kk = 0; kk < C_DIM; kk += KSTEP) {
      v16bf a = load_afrag(&sFeat[mtb * 16 + (lane & 15)][0], kk, khalf);
      int n = lane & 15;
      Frag16 f;
      f.u[0] = *(const u32x4*)(&sW2[n][kk + khalf * 16]);
      f.u[1] = *(const u32x4*)(&sW2[n][kk + khalf * 16 + 8]);
      a2 = wmma_bf16(a, f.v, a2);
    }
    int col = lane & 15;
    if (col < 3) {
      float bias = b2[col];
      #pragma unroll
      for (int r = 0; r < 8; ++r) {
        int m = mtb * 16 + khalf * 8 + r;
        sBias[m][col] = a2[r] + bias;
      }
    }
  }
  __syncthreads();

  // ---- per-row masked L1 + cosine losses ----
  if (tid < TM) {
    int row = rowBase + tid;
    float msk = (instance[row] != -1) ? 1.f : 0.f;
    float gx = centroid[row * 3 + 0] - coord[row * 3 + 0];
    float gy = centroid[row * 3 + 1] - coord[row * 3 + 1];
    float gz = centroid[row * 3 + 2] - coord[row * 3 + 2];
    float px = sBias[tid][0], py = sBias[tid][1], pz = sBias[tid][2];
    float l1 = fabsf(px - gx) + fabsf(py - gy) + fabsf(pz - gz);
    float pn = sqrtf(px * px + py * py + pz * pz) + EPSF;
    float gn = sqrtf(gx * gx + gy * gy + gz * gz) + EPSF;
    float cosv = -(px * gx + py * gy + pz * gz) / (pn * gn);
    float a0 = wave_reduce_add(l1 * msk);
    float a1 = wave_reduce_add(cosv * msk);
    float a2 = wave_reduce_add(msk);
    if (lane == 0) {
      atomicAdd(&g_acc[WS_L1],   a0);
      atomicAdd(&g_acc[WS_COS],  a1);
      atomicAdd(&g_acc[WS_MASK], a2);
    }
  }
}

// ---------------- pass4: combine scalars ----------------
__global__ void pg_pass4(const float* __restrict__ g_acc, float* __restrict__ out) {
  if (threadIdx.x == 0) {
    float seg  = g_acc[WS_SEGSUM] / (g_acc[WS_SEGCNT] + EPSF);
    float msum = g_acc[WS_MASK] + EPSF;
    float l1   = g_acc[WS_L1]  / msum;
    float cosv = g_acc[WS_COS] / msum;
    out[0] = seg + l1 + cosv;
    out[1] = seg;
    out[2] = l1;
    out[3] = cosv;
  }
}

// ---------------- launcher ----------------
extern "C" void kernel_launch(void* const* d_in, const int* in_sizes, int n_in,
                              void* d_out, int out_size, void* d_ws, size_t ws_size,
                              hipStream_t stream) {
  (void)in_sizes; (void)n_in; (void)out_size; (void)ws_size;
  const float* feat     = (const float*)d_in[0];
  const float* coord    = (const float*)d_in[1];
  const float* centroid = (const float*)d_in[2];
  const int*   segment  = (const int*)d_in[3];
  const int*   instance = (const int*)d_in[4];
  const float* w1       = (const float*)d_in[5];
  const float* b1       = (const float*)d_in[6];
  const float* gamma    = (const float*)d_in[7];
  const float* beta     = (const float*)d_in[8];
  const float* w2       = (const float*)d_in[9];
  const float* b2       = (const float*)d_in[10];
  const float* wsW      = (const float*)d_in[11];
  const float* bsB      = (const float*)d_in[12];
  char*  wsbase = (char*)d_ws;
  float* g_acc  = (float*)d_ws;
  float* out    = (float*)d_out;

  const int nBlocks = N_PTS / TM;  // 4096
  pg_init <<<1, 256, 0, stream>>>(g_acc);
  pg_prep <<<64, 256, 0, stream>>>(w1, wsW, w2, wsbase);
  pg_pass1<<<nBlocks, 256, 0, stream>>>(feat, segment, b1, bsB, wsbase);
  pg_pass2<<<1, 256, 0, stream>>>(gamma, beta, g_acc);
  pg_pass3<<<nBlocks, 256, 0, stream>>>(feat, coord, centroid, instance,
                                        b1, b2, wsbase);
  pg_pass4<<<1, 32, 0, stream>>>(g_acc, out);
}
